// AttentionDecoder_62577673503763
// MI455X (gfx1250) — compile-verified, matrix operand order checked by reading, not run on hardware
//
#include <hip/hip_runtime.h>
#include <hip/hip_bf16.h>

// ---------------------------------------------------------------------------
// AttentionDecoder for MI455X (gfx1250, wave32, WMMA).
//
//  * e_seg's ext branch is a per-batch additive constant -> cancels in
//    softmax, so w2_seg/b2_seg are never touched.
//  * Dominant GEMM (793600x256 @ 256x256, ~104 GFLOP) on
//    v_wmma_f32_16x16x32_bf16, f32 accumulation, tanh+dot(v) fused in the
//    epilogue (the [M,256] activation never hits memory).
//  * W1 pre-swizzled once into per-lane B-fragment order; per K-chunk the
//    16 KB fragment slab is DMA'd into LDS with global_load_async_to_lds_b128
//    (double-buffered, s_wait_asynccnt). B fragments are consumed from LDS in
//    two register groups of 4 (double-buffered) so the scheduler can overlap
//    ds_load latency with the 4 WMMAs of the previous group.
//  * X tiles staged through LDS per 32-wide K chunk (coalesced f32 loads ->
//    packed bf16 ds_store_b128); A fragments are two ds_load_b128 per lane.
//  * tanh uses gfx1250 v_tanh_f32 when the builtin exists.
// ---------------------------------------------------------------------------

typedef __attribute__((ext_vector_type(16))) __bf16 v16bf;
typedef __attribute__((ext_vector_type(8)))  float  v8f;

#if __has_builtin(__builtin_amdgcn_tanhf)
#define FAST_TANH(x) __builtin_amdgcn_tanhf(x)
#elif __has_builtin(__builtin_amdgcn_tanh_f32)
#define FAST_TANH(x) __builtin_amdgcn_tanh_f32(x)
#else
#define FAST_TANH(x) tanhf(x)
#endif

namespace {
constexpr int kB   = 512;
constexpr int kL   = 31;
constexpr int kS   = 50;
constexpr int kD   = 256;
constexpr int kEXT = 64;
constexpr int kLS  = kL * kS;          // 1550
constexpr float kLambda = 0.4f;
}

// ------------------------------- helpers -----------------------------------

__device__ __forceinline__ unsigned int pack_bf16_rne(float a, float b) {
  unsigned int ua = __float_as_uint(a);
  unsigned int ub = __float_as_uint(b);
  ua = (ua + 0x7FFFu + ((ua >> 16) & 1u)) >> 16;          // low half
  ub = (ub + 0x7FFFu + ((ub >> 16) & 1u)) & 0xFFFF0000u;  // high half
  return ua | ub;
}

__device__ __forceinline__ float wred_sum(float v) {
#pragma unroll
  for (int o = 16; o; o >>= 1) v += __shfl_xor(v, o, 32);
  return v;
}
__device__ __forceinline__ float wred_max(float v) {
#pragma unroll
  for (int o = 16; o; o >>= 1) v = fmaxf(v, __shfl_xor(v, o, 32));
  return v;
}
// blockDim.x == 256 (8 waves) assumed.
__device__ __forceinline__ float block_sum(float v, float* red) {
  const int lane = threadIdx.x & 31, wv = threadIdx.x >> 5;
  v = wred_sum(v);
  if (lane == 0) red[wv] = v;
  __syncthreads();
  float r = (lane < 8) ? red[lane] : 0.0f;
  r = wred_sum(r);
  __syncthreads();
  return r;
}
__device__ __forceinline__ float block_max(float v, float* red) {
  const int lane = threadIdx.x & 31, wv = threadIdx.x >> 5;
  v = wred_max(v);
  if (lane == 0) red[wv] = v;
  __syncthreads();
  float r = (lane < 8) ? red[lane] : -1e30f;
  r = wred_max(r);
  __syncthreads();
  return r;
}

// ------------------- K0: swizzle W (f32 -> bf16 fragment order) -------------
// B-matrix fragments (K x N = 32 x 16 bf16, v16bf per lane): lane =
// n%16 + 16*laneHi; lane holds K runs {0..7,16..23} (laneHi=0) or
// {8..15,24..31} (laneHi=1); frag element e = 2*j + hilo.
// Fragment fi = (kc*16+nt) occupies 512 u16; lane slice = 16 u16 (32 B).
__global__ void swizzle_w_kernel(const float* __restrict__ w,
                                 unsigned short* __restrict__ wswz) {
  const int idx = blockIdx.x * blockDim.x + threadIdx.x;   // 0 .. 65535
  if (idx >= kD * kD) return;
  const int k = idx >> 8, n = idx & 255;                   // w[k][n], row-major
  const unsigned int bits = __float_as_uint(w[idx]);
  const unsigned short h =
      (unsigned short)((bits + 0x7FFFu + ((bits >> 16) & 1u)) >> 16);
  const int kc = k >> 5, kl = k & 31, nt = n >> 4, nl = n & 15;
  const int laneHi = (kl >> 3) & 1;
  const int kk = kl - 8 * laneHi;                          // {0..7} U {16..23}
  const int j = (kk < 8) ? (kk >> 1) : (4 + ((kk - 16) >> 1));
  const int hilo = kk & 1;
  const int lane = nl + 16 * laneHi;
  const int fi = kc * 16 + nt;
  wswz[fi * 512 + lane * 16 + j * 2 + hilo] = h;
}

// -------- K1: cb_link[b][d] = ext@w2_link + b2_link + b1_link ---------------
__global__ void ext_bias_kernel(const float* __restrict__ ext,
                                const float* __restrict__ w2,
                                const float* __restrict__ b2,
                                const float* __restrict__ b1,
                                float* __restrict__ cb) {
  const int b = blockIdx.x, d = threadIdx.x;
  const float* er = ext + b * kEXT;
  float acc = b2[d] + b1[d];
#pragma unroll 8
  for (int e = 0; e < kEXT; ++e) acc += er[e] * w2[e * kD + d];
  cb[b * kD + d] = acc;
}

// -------- K2: fused  e[row] = tanh(X@W + bias(row)) . v  (bf16 WMMA) --------
// Block: 256 threads = 8 waves, 128 rows. Each wave: 16 rows x 256 cols,
// 16 C tiles, K looped in 8 chunks of 32.
// LDS: 8 KB X tile + 2 x 16 KB double-buffered W fragment slab (async DMA).
__global__ __launch_bounds__(256) void gemm_tanh_dot_kernel(
    const float* __restrict__ X, int Mtot,
    const unsigned short* __restrict__ Wswz,
    const float* __restrict__ bias, int bias_batch_stride, int rows_per_batch,
    const float* __restrict__ vvec,
    float* __restrict__ e_out) {
  __shared__ __align__(16) unsigned short xs[128 * 32];    // 8 KB bf16 X tile
  __shared__ __align__(16) char wsh[2 * 16384];            // 32 KB W frags

  const int lane = threadIdx.x & 31;
  const int wv   = threadIdx.x >> 5;                       // wave 0..7
  const int row0 = blockIdx.x * 128;
  const int t    = threadIdx.x;

  union FU { uint4 q[2]; v16bf v; };

  v8f c[16];
  const v8f z = {0.f, 0.f, 0.f, 0.f, 0.f, 0.f, 0.f, 0.f};
#pragma unroll
  for (int nt = 0; nt < 16; ++nt) c[nt] = z;

  const int stg_r    = t >> 1;                             // 0..127
  const int stg_half = t & 1;                              // 16-float half
  const int stg_row  = row0 + stg_r;

  const int m  = lane & 15;
  const int hi = lane >> 4;
  const uint4* const lq = (const uint4*)xs;

  // ---- async DMA of one 16 KB W fragment slab (K chunk) into LDS ----------
  // 256 threads x 4 x b128 = 16 KB. Generic LDS pointer's low 32 bits are the
  // hardware LDS byte address (aperture in the high bits).
  auto issue_w_copy = [&](int kc, int buf) {
    const unsigned lds0 =
        (unsigned)(uintptr_t)(wsh + buf * 16384) + (unsigned)(t * 16);
    const char* g0 = (const char*)Wswz + kc * 16384 + t * 16;
#pragma unroll
    for (int i = 0; i < 4; ++i) {
      const unsigned loff = lds0 + i * 4096;
      const unsigned long long gaddr =
          (unsigned long long)(uintptr_t)(g0 + i * 4096);
      asm volatile("global_load_async_to_lds_b128 %0, %1, off"
                   :: "v"(loff), "v"(gaddr) : "memory");
    }
  };

  issue_w_copy(0, 0);                                      // preload chunk 0

  for (int kc = 0; kc < 8; ++kc) {
    // ---- stage X[row0..row0+128, kc*32..+32) as bf16 into LDS ----
    __syncthreads();   // previous iteration finished consuming xs
    {
      float4 f0 = {0,0,0,0}, f1 = {0,0,0,0}, f2 = {0,0,0,0}, f3 = {0,0,0,0};
      if (stg_row < Mtot) {
        const float4* xp =
            (const float4*)(X + (size_t)stg_row * kD + kc * 32 + stg_half * 16);
        f0 = xp[0]; f1 = xp[1]; f2 = xp[2]; f3 = xp[3];
        if (kc + 1 < 8)  // hint next K chunk -> global_prefetch_b8
          __builtin_prefetch(X + (size_t)stg_row * kD + (kc + 1) * 32 +
                                 stg_half * 16, 0, 1);
      }
      uint4 q0, q1;
      q0.x = pack_bf16_rne(f0.x, f0.y); q0.y = pack_bf16_rne(f0.z, f0.w);
      q0.z = pack_bf16_rne(f1.x, f1.y); q0.w = pack_bf16_rne(f1.z, f1.w);
      q1.x = pack_bf16_rne(f2.x, f2.y); q1.y = pack_bf16_rne(f2.z, f2.w);
      q1.z = pack_bf16_rne(f3.x, f3.y); q1.w = pack_bf16_rne(f3.z, f3.w);
      uint4* ls = (uint4*)xs;
      ls[stg_r * 4 + stg_half * 2]     = q0;
      ls[stg_r * 4 + stg_half * 2 + 1] = q1;
    }

    // Kick off DMA of next W chunk, then wait for the *current* chunk
    // (async ops complete in order: <=4 outstanding => chunk kc landed).
    if (kc < 7) {
      issue_w_copy(kc + 1, (kc + 1) & 1);
      asm volatile("s_wait_asynccnt 0x4" ::: "memory");
    } else {
      asm volatile("s_wait_asynccnt 0x0" ::: "memory");
    }
    __syncthreads();   // xs stores + this wave's W slice visible block-wide

    // ---- A fragment: 16-bit A 16x32 layout, two ds_load_b128 per lane ----
    FU au;
    const int rowu = wv * 16 + m;
    au.q[0] = lq[rowu * 4 + hi];
    au.q[1] = lq[rowu * 4 + 2 + hi];

    // ---- 16 WMMAs across N=256; B frags from LDS in double-buffered
    //      groups of 4 so ds latency overlaps the previous group's WMMAs ----
    const uint4* wl = (const uint4*)(wsh + (kc & 1) * 16384);
    FU bg[2][4];
#pragma unroll
    for (int j = 0; j < 4; ++j) {
      bg[0][j].q[0] = wl[j * 64 + lane * 2];
      bg[0][j].q[1] = wl[j * 64 + lane * 2 + 1];
    }
#pragma unroll
    for (int g = 0; g < 4; ++g) {
      const int cur = g & 1, nxt = cur ^ 1;
      if (g < 3) {
#pragma unroll
        for (int j = 0; j < 4; ++j) {
          bg[nxt][j].q[0] = wl[((g + 1) * 4 + j) * 64 + lane * 2];
          bg[nxt][j].q[1] = wl[((g + 1) * 4 + j) * 64 + lane * 2 + 1];
        }
      }
#pragma unroll
      for (int j = 0; j < 4; ++j) {
        c[g * 4 + j] = __builtin_amdgcn_wmma_f32_16x16x32_bf16(
            false, au.v, false, bg[cur][j].v, (short)0, c[g * 4 + j],
            false, false);
      }
    }
  }

  // ---- fused epilogue: e[row] = sum_n tanh(C[row][n] + bias[n]) * v[n] ----
  // C layout: VGPR r -> row r (lanes 0-15) / row 8+r (lanes 16-31), col=lane%16.
  const int col0 = lane & 15;
  float partial[8];
#pragma unroll
  for (int r = 0; r < 8; ++r) partial[r] = 0.0f;

#pragma unroll
  for (int nt = 0; nt < 16; ++nt) {
    const int col = nt * 16 + col0;
    const float vn = vvec[col];
#pragma unroll
    for (int r = 0; r < 8; ++r) {
      const int row = row0 + wv * 16 + hi * 8 + r;
      if (row < Mtot) {
        const int batch = row / rows_per_batch;
        const float bb = bias[batch * bias_batch_stride + col];
        partial[r] += FAST_TANH(c[nt][r] + bb) * vn;
      }
    }
  }
  // reduce over the 16 lanes sharing each row (xor masks < 16 stay in half)
#pragma unroll
  for (int r = 0; r < 8; ++r) {
#pragma unroll
    for (int o = 1; o < 16; o <<= 1) partial[r] += __shfl_xor(partial[r], o, 32);
  }
  if (col0 == 0) {
#pragma unroll
    for (int r = 0; r < 8; ++r) {
      const int row = row0 + wv * 16 + hi * 8 + r;
      if (row < Mtot) e_out[row] = partial[r];
    }
  }
}

// ---------------- K3: softmax over L=31 per batch (one wave) ----------------
__global__ void link_softmax_kernel(const float* __restrict__ e_link,
                                    float* __restrict__ p_link) {
  const int b = blockIdx.x, t = threadIdx.x;
  const float x = (t < kL) ? e_link[b * kL + t] : -1e30f;
  const float mx = wred_max(x);
  const float ex = (t < kL) ? __expf(x - mx) : 0.0f;
  const float s = wred_sum(ex);
  if (t < kL) p_link[b * kL + t] = ex / s;
}

// ---- K4: softmax(e_seg) -> guide = p_seg*p_link*mask -> softmax(guide) -----
__global__ __launch_bounds__(256) void seg_guide_kernel(
    const float* __restrict__ e_seg, const float* __restrict__ p_link,
    const int* __restrict__ mask, float* __restrict__ mdist) {
  __shared__ float sg[kLS];
  __shared__ float red[8];
  const int b = blockIdx.x, t = threadIdx.x;
  const float* e = e_seg + b * kLS;

  float mx = -1e30f;
  for (int i = t; i < kLS; i += 256) mx = fmaxf(mx, e[i]);
  mx = block_max(mx, red);

  float s = 0.0f;
  for (int i = t; i < kLS; i += 256) { float ex = __expf(e[i] - mx); sg[i] = ex; s += ex; }
  s = block_sum(s, red);
  const float inv = 1.0f / s;

  float m2 = -1e30f;
  for (int i = t; i < kLS; i += 256) {
    const int l = i / kS;
    const float g = sg[i] * inv * p_link[b * kL + l] * (float)mask[b * kLS + i];
    sg[i] = g;
    m2 = fmaxf(m2, g);
  }
  m2 = block_max(m2, red);

  float s2 = 0.0f;
  for (int i = t; i < kLS; i += 256) { float ex = __expf(sg[i] - m2); sg[i] = ex; s2 += ex; }
  s2 = block_sum(s2, red);
  const float inv2 = 1.0f / s2;
  for (int i = t; i < kLS; i += 256) mdist[b * kLS + i] = sg[i] * inv2;
}

// ------ K5: att_seg/att_link weighted sums + blend + final linear -----------
__global__ __launch_bounds__(256) void weighted_final_kernel(
    const float* __restrict__ seg_feat, const float* __restrict__ link_feat,
    const float* __restrict__ mdist, const float* __restrict__ p_link,
    const float* __restrict__ lin_w, const float* __restrict__ lin_b,
    float* __restrict__ out) {
  __shared__ float wseg[kLS];
  __shared__ float wlink[kL];
  __shared__ float red[8];
  const int b = blockIdx.x, d = threadIdx.x;
  for (int i = d; i < kLS; i += 256) wseg[i] = mdist[b * kLS + i];
  if (d < kL) wlink[d] = p_link[b * kL + d];
  __syncthreads();

  const float* sf = seg_feat + (size_t)b * kLS * kD;
  float acc = 0.0f;
#pragma unroll 4
  for (int i = 0; i < kLS; ++i) acc += wseg[i] * sf[(size_t)i * kD + d];

  const float* lf = link_feat + (size_t)b * kL * kD;
  float accl = 0.0f;
#pragma unroll
  for (int l = 0; l < kL; ++l) accl += wlink[l] * lf[l * kD + d];

  const float R = (1.0f - kLambda) * acc + kLambda * accl;
  const float partial = R * lin_w[d];          // lin_w is [D,1]
  const float total = block_sum(partial, red);
  if (d == 0) out[b] = total + lin_b[0];
}

// ---------------------------------------------------------------------------

extern "C" void kernel_launch(void* const* d_in, const int* in_sizes, int n_in,
                              void* d_out, int out_size, void* d_ws,
                              size_t ws_size, hipStream_t stream) {
  (void)in_sizes; (void)n_in; (void)out_size; (void)ws_size;
  const float* seg_feat  = (const float*)d_in[0];   // [B,L,S,D]
  const float* link_feat = (const float*)d_in[1];   // [B,L,D]
  const float* ext       = (const float*)d_in[2];   // [B,EXT]
  const int*   mask      = (const int*)d_in[3];     // [B,L*S]
  const float* w1_seg    = (const float*)d_in[4];   // [D,D]
  const float* b1_seg    = (const float*)d_in[5];   // [D]
  // d_in[6]=w2_seg, d_in[7]=b2_seg: UNUSED -- per-batch constant shift of
  // e_seg, cancelled by softmax.
  const float* v_seg     = (const float*)d_in[8];   // [D]
  const float* w1_link   = (const float*)d_in[9];   // [D,D]
  const float* b1_link   = (const float*)d_in[10];  // [D]
  const float* w2_link   = (const float*)d_in[11];  // [EXT,D]
  const float* b2_link   = (const float*)d_in[12];  // [D]
  const float* v_link    = (const float*)d_in[13];  // [D]
  const float* lin_w     = (const float*)d_in[14];  // [D,1]
  const float* lin_b     = (const float*)d_in[15];  // [1]
  float* out = (float*)d_out;                       // [B,1]

  char* ws = (char*)d_ws;
  size_t off = 0;
  auto alloc = [&](size_t bytes) -> void* {
    void* p = ws + off;
    off += (bytes + 255) & ~(size_t)255;
    return p;
  };
  unsigned short* wswz_seg  = (unsigned short*)alloc((size_t)kD * kD * 2);
  unsigned short* wswz_link = (unsigned short*)alloc((size_t)kD * kD * 2);
  float* cb_link = (float*)alloc((size_t)kB * kD * sizeof(float));
  float* e_seg   = (float*)alloc((size_t)kB * kLS * sizeof(float));
  float* e_link  = (float*)alloc((size_t)kB * kL * sizeof(float));
  float* p_link  = (float*)alloc((size_t)kB * kL * sizeof(float));
  float* mdist   = (float*)alloc((size_t)kB * kLS * sizeof(float));

  // K0: weight swizzle (once per call; 128 KB each, stays L2-resident)
  swizzle_w_kernel<<<256, 256, 0, stream>>>(w1_seg, wswz_seg);
  swizzle_w_kernel<<<256, 256, 0, stream>>>(w1_link, wswz_link);

  // K1: combined link bias (ext@w2_link + b2_link + b1_link)
  ext_bias_kernel<<<kB, kD, 0, stream>>>(ext, w2_link, b2_link, b1_link, cb_link);

  // K2a: e_seg (the 104 GFLOP GEMM, fused tanh+dot) -- 6200 blocks
  const int Mseg = kB * kLS;
  gemm_tanh_dot_kernel<<<(Mseg + 127) / 128, 256, 0, stream>>>(
      seg_feat, Mseg, wswz_seg, b1_seg, /*bias_batch_stride=*/0,
      /*rows_per_batch=*/kLS, v_seg, e_seg);

  // K2b: e_link (per-batch ext bias inside tanh) -- 124 blocks
  const int Mlink = kB * kL;
  gemm_tanh_dot_kernel<<<(Mlink + 127) / 128, 256, 0, stream>>>(
      link_feat, Mlink, wswz_link, cb_link, /*bias_batch_stride=*/kD,
      /*rows_per_batch=*/kL, v_link, e_link);

  // K3: link softmax
  link_softmax_kernel<<<kB, 32, 0, stream>>>(e_link, p_link);

  // K4: seg softmax -> guide*mask -> masked softmax
  seg_guide_kernel<<<kB, 256, 0, stream>>>(e_seg, p_link, mask, mdist);

  // K5: weighted sums (second streaming pass over seg_feat) + final linear
  weighted_final_kernel<<<kB, 256, 0, stream>>>(seg_feat, link_feat, mdist,
                                                p_link, lin_w, lin_b, out);
}
